// Head_27616639713542
// MI455X (gfx1250) — compile-verified, matrix operand order checked by reading, not run on hardware
//
#include <hip/hip_runtime.h>

typedef __attribute__((ext_vector_type(16))) _Float16 v16h;
typedef __attribute__((ext_vector_type(8)))  _Float16 v8h;
typedef __attribute__((ext_vector_type(8)))  float    v8f;

#define WMMA_F16(a, b, c) \
    __builtin_amdgcn_wmma_f32_16x16x32_f16(false, (a), false, (b), (short)0, (c), false, false)

static constexpr int BATCH = 4;
static constexpr int T     = 4096;
static constexpr int C     = 1024;
static constexpr int H     = 128;
static constexpr float NEG_INF = -1e30f;
static constexpr float SCALE   = 0.03125f;  // C ** -0.5 = 1/32

__device__ __forceinline__ v16h combine16(v8h lo, v8h hi) {
    v16h r;
#pragma unroll
    for (int t = 0; t < 8; ++t) { r[t] = lo[t]; r[t + 8] = hi[t]; }
    return r;
}

// ---- CDNA5 async copy: global -> LDS, tracked by ASYNCcnt ------------------
__device__ __forceinline__ void async_b128(const _Float16* lds_p, const _Float16* g_p) {
    // LDS aperture: low 32 bits of the flat address are the LDS byte offset
    const uint32_t l = (uint32_t)(uintptr_t)lds_p;
    const uint64_t g = (uint64_t)(uintptr_t)g_p;
    asm volatile("global_load_async_to_lds_b128 %0, %1, off"
                 :: "v"(l), "v"(g) : "memory");
}
__device__ __forceinline__ void wait_async0() {
    asm volatile("s_wait_asynccnt 0x0" ::: "memory");
}

// Cooperative async copy of a 128-row x 128-half tile (32 KB) into LDS.
// 256 threads x 8 chunks x 16 B.  gstride = global row stride in halves.
__device__ __forceinline__ void load_tile_async(_Float16* lds_base,
                                                const _Float16* gbase,
                                                int gstride, int tid) {
#pragma unroll
    for (int c = 0; c < 8; ++c) {
        const int chunk = tid + c * 256;          // 0..2047
        const int row   = chunk >> 4;
        const int off   = (chunk & 15) * 8;       // halves
        async_b128(lds_base + row * 128 + off,
                   gbase + (size_t)row * gstride + off);
    }
}

// ---------------------------------------------------------------------------
// Prep 0a: x (f32) -> xh (f16), 8 elements/thread, pure b128 traffic.
// ---------------------------------------------------------------------------
__global__ __launch_bounds__(256)
void cvt_x_kernel(const float* __restrict__ x, _Float16* __restrict__ xh) {
    const size_t i = ((size_t)blockIdx.x * 256 + threadIdx.x) * 8;
    const float4* xp = (const float4*)(x + i);
    const float4 a = xp[0], b = xp[1];
    v8h o;
    o[0] = (_Float16)a.x; o[1] = (_Float16)a.y; o[2] = (_Float16)a.z; o[3] = (_Float16)a.w;
    o[4] = (_Float16)b.x; o[5] = (_Float16)b.y; o[6] = (_Float16)b.z; o[7] = (_Float16)b.w;
    *(v8h*)(xh + i) = o;
}

// ---------------------------------------------------------------------------
// Prep 0b: W[C][H] f32 -> Wt[which][H][C] f16 (transposed for contiguous
// B-fragments in the projection GEMM).  Tiny: 0.75 MB total.
// ---------------------------------------------------------------------------
__global__ __launch_bounds__(128)
void cvt_w_kernel(const float* __restrict__ Wk, const float* __restrict__ Wq,
                  const float* __restrict__ Wv, _Float16* __restrict__ Wt) {
    const int which = blockIdx.y;
    const float* W  = (which == 0) ? Wk : (which == 1) ? Wq : Wv;
    const int h = blockIdx.x;
    _Float16* o = Wt + ((size_t)which * H + h) * C;
    for (int c = threadIdx.x; c < C; c += 128)
        o[c] = (_Float16)W[(size_t)c * H + h];
}

// ---------------------------------------------------------------------------
// Kernel 1: K/Q/V projections.  Each wave = one 16x16 tile, 32 WMMAs over
// C=1024, all-b128 fragment loads.  K,Q row-major; V stored as Vt[b][h][t].
// ---------------------------------------------------------------------------
__global__ __launch_bounds__(256)
void qkv_proj_kernel(const _Float16* __restrict__ xh,
                     const _Float16* __restrict__ Wt,
                     _Float16* __restrict__ Kh,
                     _Float16* __restrict__ Qh,
                     _Float16* __restrict__ Vt) {
    const int lane  = threadIdx.x & 31;
    const int wave  = threadIdx.x >> 5;
    const int mt    = blockIdx.x;
    const int which = blockIdx.y;

    const int r     = lane & 15;
    const int half8 = (lane >> 4) * 8;
    const int row   = mt * 16 + r;
    const int col   = wave * 16 + r;

    const _Float16* xrow  = xh + (size_t)row * C;
    const _Float16* wbase = Wt + ((size_t)which * H + col) * C;

    v8f acc = {};
    for (int kk = 0; kk < C; kk += 32) {
        const _Float16* xp = xrow + kk + half8;
        const _Float16* wp = wbase + kk + half8;
        v16h a = combine16(*(const v8h*)xp, *(const v8h*)(xp + 16));
        v16h b = combine16(*(const v8h*)wp, *(const v8h*)(wp + 16));
        acc = WMMA_F16(a, b, acc);
    }

    if (which == 2) {
        const int token = mt * 16 + half8;
        const int bidx  = token / T;
        const int tloc  = token % T;
        v8h o;
#pragma unroll
        for (int t = 0; t < 8; ++t) o[t] = (_Float16)acc[t];
        *(v8h*)(Vt + ((size_t)bidx * H + col) * T + tloc) = o;
    } else {
        _Float16* Out = (which == 0) ? Kh : Qh;
#pragma unroll
        for (int t = 0; t < 8; ++t)
            Out[(size_t)(mt * 16 + half8 + t) * H + col] = (_Float16)acc[t];
    }
}

// ---------------------------------------------------------------------------
// Kernel 2: flash attention with async-LDS software pipeline.
//   wei[i,j] = (k_i . q_j)/32, causal j<=i, softmax_j, out = wei @ v.
// Block = 8 waves = 128 i-rows.  Per j-tile:
//   [V_j async in flight] S-phase from LDS Q_j -> softmax -> P to LDS
//   wait_async + barrier  [Q_{j+1} async in flight] PV-phase from LDS V_j
//   wait_async + barrier.
// ---------------------------------------------------------------------------
__global__ __launch_bounds__(256)
void attn_kernel(const _Float16* __restrict__ Kh,
                 const _Float16* __restrict__ Qh,
                 const _Float16* __restrict__ Vt,
                 float* __restrict__ out) {
    __shared__ __align__(16) _Float16 Qt[128 * 128];     // 32 KB: q rows (j-tile)
    __shared__ __align__(16) _Float16 Vtile[128 * 128];  // 32 KB: Vt rows (h-major)
    __shared__ __align__(16) _Float16 Pbuf[8][16 * 128]; // 32 KB: per-wave P staging

    const int tid   = threadIdx.x;
    const int lane  = tid & 31;
    const int wave  = tid >> 5;
    const int b     = blockIdx.y;
    const int itile = blockIdx.x;
    const int i0    = itile * 128 + wave * 16;
    const int jmax  = itile * 128;

    const _Float16* Kb  = Kh + (size_t)b * T * H;
    const _Float16* Qb  = Qh + (size_t)b * T * H;
    const _Float16* VtB = Vt + (size_t)b * H * T;

    const int r     = lane & 15;
    const int half8 = (lane >> 4) * 8;

    // K A-fragments are invariant across j-tiles: hoist into registers.
    v16h afrag[4];
#pragma unroll
    for (int hk = 0; hk < 4; ++hk) {
        const _Float16* kp = Kb + (size_t)(i0 + r) * H + hk * 32 + half8;
        afrag[hk] = combine16(*(const v8h*)kp, *(const v8h*)(kp + 16));
    }

    v8f   acc[8];
    float m[8], l[8];
#pragma unroll
    for (int h = 0; h < 8; ++h) acc[h] = (v8f){};
#pragma unroll
    for (int t = 0; t < 8; ++t) { m[t] = NEG_INF; l[t] = 0.f; }

    _Float16* pb = &Pbuf[wave][0];

    // Prologue: bring in Q-tile for j0 = 0.
    load_tile_async(Qt, Qb, H, tid);
    wait_async0();
    __syncthreads();

    for (int j0 = 0; j0 <= jmax; j0 += 128) {
        // V_j tile DMA overlaps the S-phase below.
        load_tile_async(Vtile, VtB + j0, T, tid);

        // L2 prefetch for the tile after next (global_prefetch_b8).
        if (j0 + 256 <= jmax) {
            const int prr = tid & 127;
            __builtin_prefetch(Qb + (size_t)(j0 + 256 + prr) * H, 0, 1);
            __builtin_prefetch(VtB + (size_t)prr * T + j0 + 256, 0, 1);
        }

        // ---- S = K_i x Q_j^T from LDS Q-tile ----
        v8f s[8];
#pragma unroll
        for (int f = 0; f < 8; ++f) s[f] = (v8f){};
#pragma unroll
        for (int hk = 0; hk < 4; ++hk) {
            const v16h a = afrag[hk];
#pragma unroll
            for (int f = 0; f < 8; ++f) {
                const _Float16* qp = Qt + (f * 16 + r) * 128 + hk * 32 + half8;
                v16h bq = combine16(*(const v8h*)qp, *(const v8h*)(qp + 16));
                s[f] = WMMA_F16(a, bq, s[f]);
            }
        }

        // ---- scale + causal mask (diagonal j-tile only) ----
        const bool diag = (j0 == jmax);
#pragma unroll
        for (int f = 0; f < 8; ++f) {
#pragma unroll
            for (int t = 0; t < 8; ++t) {
                float v = s[f][t] * SCALE;
                if (diag) {
                    const int ir = i0 + half8 + t;
                    const int jc = j0 + f * 16 + r;
                    if (jc > ir) v = NEG_INF;
                }
                s[f][t] = v;
            }
        }

        // ---- online softmax (rows in VGPR index, cols across 16 lanes) ----
#pragma unroll
        for (int t = 0; t < 8; ++t) {
            float rmax = NEG_INF;
#pragma unroll
            for (int f = 0; f < 8; ++f) rmax = fmaxf(rmax, s[f][t]);
#pragma unroll
            for (int off = 1; off < 16; off <<= 1)
                rmax = fmaxf(rmax, __shfl_xor(rmax, off, 32));
            const float mnew  = fmaxf(m[t], rmax);
            const float alpha = __expf(m[t] - mnew);
            float rsum = 0.f;
#pragma unroll
            for (int f = 0; f < 8; ++f) {
                const float p = __expf(s[f][t] - mnew);
                s[f][t] = p;
                rsum += p;
            }
#pragma unroll
            for (int off = 1; off < 16; off <<= 1)
                rsum += __shfl_xor(rsum, off, 32);
            l[t] = l[t] * alpha + rsum;
            m[t] = mnew;
#pragma unroll
            for (int h = 0; h < 8; ++h) acc[h][t] *= alpha;
        }

        // ---- P (D-layout) -> per-wave LDS (A-layout source) ----
#pragma unroll
        for (int f = 0; f < 8; ++f)
#pragma unroll
            for (int t = 0; t < 8; ++t)
                pb[(half8 + t) * 128 + f * 16 + r] = (_Float16)s[f][t];
        asm volatile("s_wait_dscnt 0" ::: "memory");  // LDS in-order per wave

        // V_j present for this wave; barrier: present for all, Q_j consumed.
        wait_async0();
        __syncthreads();

        // Next Q-tile DMA overlaps the PV-phase below.
        if (j0 + 128 <= jmax)
            load_tile_async(Qt, Qb + (size_t)(j0 + 128) * H, H, tid);

        // ---- O += P x V from LDS V-tile ----
#pragma unroll
        for (int jk = 0; jk < 128; jk += 32) {
            const _Float16* pr = pb + r * 128 + jk + half8;
            v16h a = combine16(*(const v8h*)pr, *(const v8h*)(pr + 16));
#pragma unroll
            for (int h = 0; h < 8; ++h) {
                const _Float16* vp = Vtile + (h * 16 + r) * 128 + jk + half8;
                v16h bv = combine16(*(const v8h*)vp, *(const v8h*)(vp + 16));
                acc[h] = WMMA_F16(a, bv, acc[h]);
            }
        }

        // Q_{j+1} present; barrier: V_j consumed by all before next overwrite.
        wait_async0();
        __syncthreads();
    }

    // ---- normalize and write fp32 output ----
    float* ob = out + (size_t)b * T * H;
#pragma unroll
    for (int t = 0; t < 8; ++t) {
        const float invl = 1.0f / l[t];
#pragma unroll
        for (int h = 0; h < 8; ++h)
            ob[(size_t)(i0 + half8 + t) * H + h * 16 + r] = acc[h][t] * invl;
    }
}

// ---------------------------------------------------------------------------
extern "C" void kernel_launch(void* const* d_in, const int* in_sizes, int n_in,
                              void* d_out, int out_size, void* d_ws, size_t ws_size,
                              hipStream_t stream) {
    const float* x  = (const float*)d_in[0];
    const float* Wk = (const float*)d_in[1];
    const float* Wq = (const float*)d_in[2];
    const float* Wv = (const float*)d_in[3];

    const size_t MC  = (size_t)BATCH * T * C;
    const size_t BTH = (size_t)BATCH * T * H;

    _Float16* xh = (_Float16*)d_ws;              // 32 MB
    _Float16* Wt = xh + MC;                      // 0.75 MB
    _Float16* Kh = Wt + (size_t)3 * H * C;       // 4 MB
    _Float16* Qh = Kh + BTH;                     // 4 MB
    _Float16* Vt = Qh + BTH;                     // 4 MB  ([b][h][t])

    cvt_x_kernel<<<dim3(MC / (256 * 8)), 256, 0, stream>>>(x, xh);
    cvt_w_kernel<<<dim3(H, 3), 128, 0, stream>>>(Wk, Wq, Wv, Wt);

    qkv_proj_kernel<<<dim3((BATCH * T) / 16, 3), 256, 0, stream>>>(
        xh, Wt, Kh, Qh, Vt);

    attn_kernel<<<dim3(T / 128, BATCH), 256, 0, stream>>>(Kh, Qh, Vt, (float*)d_out);
}
